// NodalMLP_54889682043442
// MI455X (gfx1250) — matrix operand order repaired and checked by reading, not compiled, not run on hardware
//
#include <hip/hip_runtime.h>
#include <hip/hip_bf16.h>

typedef __attribute__((ext_vector_type(16))) _Float16 v16h;
typedef __attribute__((ext_vector_type(8)))  _Float16 v8h;
typedef __attribute__((ext_vector_type(8)))  float    v8f;

#define N_NODES 400
#define HID 96
#define DECW 64
#define FEAT_STATIC 8
#define FEAT (FEAT_STATIC + N_NODES)   // 408
#define BATCH 16
#define NEDGE (N_NODES*(N_NODES-1)/2)  // 79800
#define ET32 ((NEDGE + 31)/32)         // 2494 tiles of 32 edges
#define EPSV 1e-8f

// ---- workspace layout (bytes) ----
#define WS_INVMEAN 0
#define WS_H16     256
#define WS_H16_BYTES (BATCH*N_NODES*HID*2)      // 1,228,800
#define WS_PW      (WS_H16 + WS_H16_BYTES)      // packed W_d1 (36 frags) + W_d2 (8 frags)

// ---------------- per-batch mean of sc_matrix ----------------
__global__ void mean_kernel(const float* __restrict__ sc, float* __restrict__ inv_mean) {
  __shared__ float red[256];
  int b = blockIdx.x;
  const float* p = sc + (size_t)b * N_NODES * N_NODES;
  float s = 0.f;
  for (int i = threadIdx.x; i < N_NODES * N_NODES; i += 256) s += p[i];
  red[threadIdx.x] = s;
  __syncthreads();
  for (int off = 128; off > 0; off >>= 1) {
    if (threadIdx.x < off) red[threadIdx.x] += red[threadIdx.x + off];
    __syncthreads();
  }
  if (threadIdx.x == 0) {
    float m = red[0] / (float)(N_NODES * N_NODES);
    inv_mean[b] = 1.f / fmaxf(m, EPSV);
  }
}

// -------- pack decoder weights into WMMA B-fragment order (f16) --------
// frag f = kt*4+nt ; element idx = f*512 + lane*16 + s
// lane<16 : N = nt*16+lane,    K = kt*32 + s
// lane>=16: N = nt*16+lane-16, K = kt*32 + 16 + s
__global__ void pack_weights_kernel(const float* __restrict__ Wd1,
                                    const float* __restrict__ Wd2,
                                    _Float16* __restrict__ p1,
                                    _Float16* __restrict__ p2) {
  for (int idx = threadIdx.x; idx < 36 * 512; idx += 256) {
    int f = idx >> 9, r = idx & 511;
    int lane = r >> 4, s = r & 15;
    int kt = f >> 2, nt = f & 3;
    int N = nt * 16 + (lane & 15);
    int K = kt * 32 + ((lane < 16) ? s : (16 + s));
    p1[idx] = (_Float16)Wd1[K * DECW + N];
  }
  for (int idx = threadIdx.x; idx < 8 * 512; idx += 256) {
    int f = idx >> 9, r = idx & 511;
    int lane = r >> 4, s = r & 15;
    int kt = f >> 2, nt = f & 3;
    int N = nt * 16 + (lane & 15);
    int K = kt * 32 + ((lane < 16) ? s : (16 + s));
    p2[idx] = (_Float16)Wd2[K * DECW + N];
  }
}

// ---------------- encoder: h = prelu(prelu(feats@We1+b)@We2+b) ----------------
__global__ void encoder_kernel(const float* __restrict__ nf, const float* __restrict__ sc,
                               const float* __restrict__ smean, const float* __restrict__ sstd,
                               const float* __restrict__ We1, const float* __restrict__ be1,
                               const float* __restrict__ ae1,
                               const float* __restrict__ We2, const float* __restrict__ be2,
                               const float* __restrict__ ae2,
                               const float* __restrict__ inv_mean,
                               _Float16* __restrict__ h16) {
  __shared__ float feats[FEAT];
  __shared__ float h1[HID];
  int b = blockIdx.x / N_NODES;
  int node = blockIdx.x % N_NODES;
  int t = threadIdx.x;  // 0..95
  float inv = inv_mean[b];
  if (t < FEAT_STATIC)
    feats[t] = (nf[(b * N_NODES + node) * FEAT_STATIC + t] - smean[t]) / (sstd[t] + EPSV);
  const float* scrow = sc + ((size_t)b * N_NODES + node) * N_NODES;
  for (int j = t; j < N_NODES; j += HID) feats[FEAT_STATIC + j] = scrow[j] * inv;
  __syncthreads();
  float a1 = ae1[0];
  float acc = be1[t];
  for (int k = 0; k < FEAT; ++k) acc += feats[k] * We1[k * HID + t];
  acc = acc >= 0.f ? acc : a1 * acc;
  h1[t] = acc;
  __syncthreads();
  float a2 = ae2[0];
  float acc2 = be2[t];
  for (int k = 0; k < HID; ++k) acc2 += h1[k] * We2[k * HID + t];
  acc2 = acc2 >= 0.f ? acc2 : a2 * acc2;
  h16[((size_t)b * N_NODES + node) * HID + t] = (_Float16)acc2;
}

// packed-f16 edge-feature chunk: op 0=sum, 1=|diff|, 2=prod (op is compile-time)
__device__ __forceinline__ v8h chunk_op(const _Float16* __restrict__ hi,
                                        const _Float16* __restrict__ hj,
                                        int c0, int op) {
  v8h a = *(const v8h*)(hi + c0);
  v8h b = *(const v8h*)(hj + c0);
  if (op == 0) return a + b;          // v_pk_add_f16
  if (op == 2) return a * b;          // v_pk_mul_f16
  union { v8h h; unsigned u[4]; } d;  // |a-b| : pk_add(neg) + and-mask
  d.h = a - b;
#pragma unroll
  for (int k = 0; k < 4; ++k) d.u[k] &= 0x7FFF7FFFu;
  return d.h;
}

// ---------------- decoder: one wave per 32-edge tile, WMMA f16->f32 ----------------
__global__ __launch_bounds__(128)
void decoder_kernel(const _Float16* __restrict__ h16,
                    const _Float16* __restrict__ pw1,   // 36 packed frags
                    const _Float16* __restrict__ pw2,   // 8 packed frags
                    const float* __restrict__ bd1, const float* __restrict__ ad1,
                    const float* __restrict__ bd2, const float* __restrict__ ad2,
                    const float* __restrict__ Wd3, const float* __restrict__ bd3,
                    const int* __restrict__ ei, const int* __restrict__ ej,
                    float* __restrict__ out) {
  // per-wave scratch: z1 (32x64 f16, 4KB) + z2 (32x64 f32, 8KB) = 12KB; 4 waves = 48KB
  __shared__ __align__(16) _Float16 zmem[4 * 6144];

  int wv = threadIdx.x >> 5;
  int lane = threadIdx.x & 31;
  int gw = blockIdx.x * 4 + wv;
  int b = gw / ET32;
  int et = gw % ET32;
  int ebase = et * 32;

  _Float16* z1 = zmem + wv * 6144;        // [32][64] f16
  float*    z2 = (float*)(z1 + 2048);     // [32][64] f32

  int r  = lane & 15;   // row within a 16-row A tile
  int hs = lane >> 4;   // A K-chunk half-select

  // edge indices for the two M-tiles this lane feeds (garbage rows are never stored)
  int e0 = ebase + r, e1 = ebase + 16 + r;
  int c0i = ei[min(e0, NEDGE - 1)], c0j = ej[min(e0, NEDGE - 1)];
  int c1i = ei[min(e1, NEDGE - 1)], c1j = ej[min(e1, NEDGE - 1)];
  const _Float16* hb = h16 + (size_t)b * N_NODES * HID;
  const _Float16* hi0 = hb + c0i * HID;
  const _Float16* hj0 = hb + c0j * HID;
  const _Float16* hi1 = hb + c1i * HID;
  const _Float16* hj1 = hb + c1j * HID;

  // ---- decoder layer 1: edge_feat(32x288) @ Wd1(288x64) ----
  v8f acc[2][4] = {};
#pragma unroll
  for (int kt = 0; kt < 9; ++kt) {
    int op = kt / 3;                       // 0:sum 1:|diff| 2:prod (compile-time)
    int c0 = kt * 32 + 16 * hs - op * 96;  // h column of first chunk
    union { v16h v; v8h h[2]; } A0, A1;
    A0.h[0] = chunk_op(hi0, hj0, c0,      op);
    A0.h[1] = chunk_op(hi0, hj0, c0 + 8,  op);
    A1.h[0] = chunk_op(hi1, hj1, c0,      op);
    A1.h[1] = chunk_op(hi1, hj1, c0 + 8,  op);
#pragma unroll
    for (int nt = 0; nt < 4; ++nt) {
      const _Float16* bp = pw1 + ((kt * 4 + nt) * 32 + lane) * 16;
      union { v16h v; v8h h[2]; } Bf;
      Bf.h[0] = *(const v8h*)bp;
      Bf.h[1] = *(const v8h*)(bp + 8);
      acc[0][nt] = __builtin_amdgcn_wmma_f32_16x16x32_f16(
          false, A0.v, false, Bf.v, (short)0, acc[0][nt], false, false);
      acc[1][nt] = __builtin_amdgcn_wmma_f32_16x16x32_f16(
          false, A1.v, false, Bf.v, (short)0, acc[1][nt], false, false);
    }
  }

  // bias + prelu, D layout -> z1 row-major f16 via LDS
  float a1 = ad1[0];
#pragma unroll
  for (int nt = 0; nt < 4; ++nt) {
    int N = nt * 16 + r;
    float bv = bd1[N];
#pragma unroll
    for (int mt = 0; mt < 2; ++mt) {
#pragma unroll
      for (int v = 0; v < 8; ++v) {
        int M = mt * 16 + v + 8 * hs;
        float z = acc[mt][nt][v] + bv;
        z = z >= 0.f ? z : a1 * z;
        z1[M * 64 + N] = (_Float16)z;
      }
    }
  }
  asm volatile("s_wait_dscnt 0" ::: "memory");

  // ---- decoder layer 2: z1(32x64) @ Wd2(64x64) ----
  v8f acc2[2][4] = {};
#pragma unroll
  for (int kt = 0; kt < 2; ++kt) {
    union { v16h v; v8h h[2]; } A0, A1;
    const _Float16* ap0 = z1 + r * 64 + kt * 32 + hs * 16;
    const _Float16* ap1 = ap0 + 16 * 64;
    A0.h[0] = *(const v8h*)ap0;
    A0.h[1] = *(const v8h*)(ap0 + 8);
    A1.h[0] = *(const v8h*)ap1;
    A1.h[1] = *(const v8h*)(ap1 + 8);
#pragma unroll
    for (int nt = 0; nt < 4; ++nt) {
      const _Float16* bp = pw2 + ((kt * 4 + nt) * 32 + lane) * 16;
      union { v16h v; v8h h[2]; } Bf;
      Bf.h[0] = *(const v8h*)bp;
      Bf.h[1] = *(const v8h*)(bp + 8);
      acc2[0][nt] = __builtin_amdgcn_wmma_f32_16x16x32_f16(
          false, A0.v, false, Bf.v, (short)0, acc2[0][nt], false, false);
      acc2[1][nt] = __builtin_amdgcn_wmma_f32_16x16x32_f16(
          false, A1.v, false, Bf.v, (short)0, acc2[1][nt], false, false);
    }
  }

  float a2 = ad2[0];
#pragma unroll
  for (int nt = 0; nt < 4; ++nt) {
    int N = nt * 16 + r;
    float bv = bd2[N];
#pragma unroll
    for (int mt = 0; mt < 2; ++mt) {
#pragma unroll
      for (int v = 0; v < 8; ++v) {
        int M = mt * 16 + v + 8 * hs;
        float z = acc2[mt][nt][v] + bv;
        z = z >= 0.f ? z : a2 * z;
        z2[M * 64 + N] = z;
      }
    }
  }
  asm volatile("s_wait_dscnt 0" ::: "memory");

  // ---- decoder layer 3: 64 -> 1 dot, one edge per lane ----
  int e3 = ebase + lane;
  if (e3 < NEDGE) {
    float s = bd3[0];
    const float* zr = z2 + lane * 64;
#pragma unroll 8
    for (int c = 0; c < 64; ++c) s += zr[c] * Wd3[c];
    out[(size_t)b * NEDGE + e3] = s;
  }
}

extern "C" void kernel_launch(void* const* d_in, const int* in_sizes, int n_in,
                              void* d_out, int out_size, void* d_ws, size_t ws_size,
                              hipStream_t stream) {
  (void)in_sizes; (void)n_in; (void)out_size; (void)ws_size;
  const float* nf    = (const float*)d_in[1];
  const float* sc    = (const float*)d_in[2];
  const float* smean = (const float*)d_in[3];
  const float* sstd  = (const float*)d_in[4];
  const float* We1   = (const float*)d_in[5];
  const float* be1   = (const float*)d_in[6];
  const float* ae1   = (const float*)d_in[7];
  const float* We2   = (const float*)d_in[8];
  const float* be2   = (const float*)d_in[9];
  const float* ae2   = (const float*)d_in[10];
  const float* Wd1   = (const float*)d_in[11];
  const float* bd1   = (const float*)d_in[12];
  const float* ad1   = (const float*)d_in[13];
  const float* Wd2   = (const float*)d_in[14];
  const float* bd2   = (const float*)d_in[15];
  const float* ad2   = (const float*)d_in[16];
  const float* Wd3   = (const float*)d_in[17];
  const float* bd3   = (const float*)d_in[18];
  const int*   ei    = (const int*)d_in[19];
  const int*   ej    = (const int*)d_in[20];

  char* ws = (char*)d_ws;
  float*     inv_mean = (float*)(ws + WS_INVMEAN);
  _Float16*  h16      = (_Float16*)(ws + WS_H16);
  _Float16*  pw1      = (_Float16*)(ws + WS_PW);
  _Float16*  pw2      = pw1 + 36 * 512;

  mean_kernel<<<BATCH, 256, 0, stream>>>(sc, inv_mean);
  pack_weights_kernel<<<1, 256, 0, stream>>>(Wd1, Wd2, pw1, pw2);
  encoder_kernel<<<BATCH * N_NODES, HID, 0, stream>>>(
      nf, sc, smean, sstd, We1, be1, ae1, We2, be2, ae2, inv_mean, h16);
  int nblocks = (BATCH * ET32 + 3) / 4;  // 9976 blocks, 4 waves (32-edge tiles) each
  decoder_kernel<<<nblocks, 128, 0, stream>>>(
      h16, pw1, pw2, bd1, ad1, bd2, ad2, Wd3, bd3, ei, ej, (float*)d_out);
}
